// LocalGNNBranch_2070174236742
// MI455X (gfx1250) — compile-verified, matrix operand order checked by reading, not run on hardware
//
#include <hip/hip_runtime.h>
#include <hip/hip_bf16.h>

#define NN 100000
#define EE 1600000
#define BB 64
#define DD 64

typedef __attribute__((ext_vector_type(16))) __bf16 v16bf;
typedef __attribute__((ext_vector_type(8)))  float  v8f;

// ---------------- utility kernels ----------------

__global__ void zero_f32(float* __restrict__ p, long long n) {
  long long i = blockIdx.x * (long long)blockDim.x + threadIdx.x;
  long long stride = (long long)gridDim.x * blockDim.x;
  for (; i < n; i += stride) p[i] = 0.0f;
}

__global__ void degree_kernel(const int* __restrict__ dst, float* __restrict__ deg, int e) {
  int i = blockIdx.x * blockDim.x + threadIdx.x;
  if (i < e) atomicAdd(&deg[dst[i]], 1.0f);
}

__global__ void invdeg_kernel(float* __restrict__ deg, int n) {
  int i = blockIdx.x * blockDim.x + threadIdx.x;
  if (i < n) deg[i] = 1.0f / fmaxf(deg[i], 1.0f);
}

__global__ void cvt_bf16_kernel(const float* __restrict__ in, __bf16* __restrict__ out, int n) {
  int i = blockIdx.x * blockDim.x + threadIdx.x;
  if (i < n) out[i] = (__bf16)in[i];
}

// scatter-add of source-node feature rows into agg[dst] (segment_sum over edges)
__global__ void scatter_kernel(const int* __restrict__ src, const int* __restrict__ dst,
                               const float* __restrict__ x, float* __restrict__ agg,
                               long long total) {
  long long i = blockIdx.x * (long long)blockDim.x + threadIdx.x;
  long long stride = (long long)gridDim.x * blockDim.x;
  for (; i < total; i += stride) {
    long long e = i >> 6;
    int f = (int)(i & 63);
    atomicAdd(&agg[(long long)dst[e] * DD + f], x[(long long)src[e] * DD + f]);
  }
}

// final readout: segment_sum of h over batch ids -> [B, D]
__global__ void readout_kernel(const float* __restrict__ h, const int* __restrict__ batch,
                               float* __restrict__ out, long long total) {
  long long i = blockIdx.x * (long long)blockDim.x + threadIdx.x;
  long long stride = (long long)gridDim.x * blockDim.x;
  for (; i < total; i += stride) {
    long long node = i >> 6;
    int f = (int)(i & 63);
    atomicAdd(&out[(long long)batch[node] * DD + f], h[i]);
  }
}

// ---------------- WMMA fragment helpers ----------------
// 16-bit A layout (16x32, MxK): lane L -> M = L&15, half h = L>>4.
// VGPR v<4 holds K = 2v+8h, 2v+1+8h; v>=4 holds K = 16+2(v-4)+8h, +1.
// => fragment elements 0..7  = K = kb*32 + 8h + (0..7)      (contiguous)
//    fragment elements 8..15 = K = kb*32 + 8h + 16 + (0..7) (contiguous)
__device__ __forceinline__ v16bf load_a_frag(const float* __restrict__ rowp,
                                             int kb, int h, float scale) {
  v16bf a;
  const float* p0 = rowp + kb * 32 + h * 8;
  const float* p1 = p0 + 16;
#pragma unroll
  for (int i = 0; i < 8; ++i) {
    a[i]     = (__bf16)(p0[i] * scale);
    a[i + 8] = (__bf16)(p1[i] * scale);
  }
  return a;
}

// B (32x16, KxN): same K striping, lane -> N = L&15. W row-major (n, k) so the
// K-runs are contiguous in memory.
__device__ __forceinline__ v16bf load_b_frag(const __bf16* __restrict__ wrow,
                                             int kb, int h) {
  v16bf b;
  const __bf16* p0 = wrow + kb * 32 + h * 8;
  const __bf16* p1 = p0 + 16;
#pragma unroll
  for (int i = 0; i < 8; ++i) {
    b[i]     = p0[i];
    b[i + 8] = p1[i];
  }
  return b;
}

// ---------------- fused SAGE layer: mean@Wl^T + bl + x@Wr^T, L2-norm, ReLU ----------------
// One wave32 per 16-node tile; 8 waves (tiles) per 256-thread block.
__global__ __launch_bounds__(256) void sage_gemm_norm_kernel(
    const float* __restrict__ agg, const float* __restrict__ inv_deg,
    const float* __restrict__ xin, const __bf16* __restrict__ Wl,
    const __bf16* __restrict__ Wr, const float* __restrict__ bias,
    float* __restrict__ hout, int n_nodes) {
  __shared__ float lds[8][16 * DD];

  const int wave = threadIdx.x >> 5;
  const int lane = threadIdx.x & 31;
  const int tile = blockIdx.x * 8 + wave;
  const int node0 = tile * 16;
  if (node0 >= n_nodes) return;   // wave-uniform: EXEC stays all-ones for WMMA

  const int m = lane & 15;
  const int h = lane >> 4;
  const int row = node0 + m;
  const float sc = inv_deg[row];

  v8f c[4] = {};  // 4 N-tiles of the 16x64 f32 accumulator

#pragma unroll
  for (int kb = 0; kb < 2; ++kb) {       // K = 64 as two K=32 WMMA steps
    const v16bf aM = load_a_frag(agg + (long long)row * DD, kb, h, sc);   // mean row
    const v16bf aX = load_a_frag(xin + (long long)row * DD, kb, h, 1.0f); // root row
#pragma unroll
    for (int nt = 0; nt < 4; ++nt) {
      const int n = nt * 16 + m;
      const v16bf bL = load_b_frag(Wl + n * DD, kb, h);
      const v16bf bR = load_b_frag(Wr + n * DD, kb, h);
      c[nt] = __builtin_amdgcn_wmma_f32_16x16x32_bf16(
          false, aM, false, bL, (short)0, c[nt], false, false);
      c[nt] = __builtin_amdgcn_wmma_f32_16x16x32_bf16(
          false, aX, false, bR, (short)0, c[nt], false, false);
    }
  }

  // Spill C to LDS with the documented layout: VGPR r -> row r + 8h, col = nt*16 + (lane&15)
  float* myl = lds[wave];
#pragma unroll
  for (int nt = 0; nt < 4; ++nt) {
    const int col = nt * 16 + m;
    const float bv = bias[col];
#pragma unroll
    for (int r = 0; r < 8; ++r)
      myl[(r + h * 8) * DD + col] = c[nt][r] + bv;
  }

  // Wave-private LDS region: only need the wave's DS ops drained (CDNA5 split counter)
  asm volatile("s_wait_dscnt 0" ::: "memory");

  // Row-wise L2 normalize + ReLU; lanes 0..15 each own one row
  if (lane < 16) {
    const float* rp = myl + lane * DD;
    float ss = 0.0f;
#pragma unroll
    for (int f = 0; f < DD; ++f) ss += rp[f] * rp[f];
    const float inv = 1.0f / fmaxf(sqrtf(ss), 1e-12f);
    float* op = hout + (long long)(node0 + lane) * DD;
#pragma unroll
    for (int f = 0; f < DD; ++f) {
      const float v = rp[f] * inv;
      op[f] = v > 0.0f ? v : 0.0f;
    }
  }
}

// ---------------- launch ----------------

extern "C" void kernel_launch(void* const* d_in, const int* in_sizes, int n_in,
                              void* d_out, int out_size, void* d_ws, size_t ws_size,
                              hipStream_t stream) {
  const float* x_raw = (const float*)d_in[0];
  const int* edge    = (const int*)d_in[1];
  const int* src     = edge;        // edge_index row 0
  const int* dst     = edge + EE;   // edge_index row 1
  const int* batch   = (const int*)d_in[2];
  const float* Wl[3] = {(const float*)d_in[3], (const float*)d_in[6], (const float*)d_in[9]};
  const float* bl[3] = {(const float*)d_in[4], (const float*)d_in[7], (const float*)d_in[10]};
  const float* Wr[3] = {(const float*)d_in[5], (const float*)d_in[8], (const float*)d_in[11]};

  // workspace layout
  float* deg = (float*)d_ws;                       // N (becomes inv_deg)
  float* agg = deg + NN;                           // N*64
  float* hA  = agg + (long long)NN * DD;           // N*64
  float* hB  = hA + (long long)NN * DD;            // N*64
  __bf16* wbf = (__bf16*)(hB + (long long)NN * DD); // 6 * 4096 bf16 (Wl/Wr per layer)

  // deg + output accumulators must be zeroed every call (graph replays)
  zero_f32<<<128, 256, 0, stream>>>(deg, NN);
  zero_f32<<<16, 256, 0, stream>>>((float*)d_out, (long long)BB * DD);

  degree_kernel<<<(EE + 255) / 256, 256, 0, stream>>>(dst, deg, EE);
  invdeg_kernel<<<(NN + 255) / 256, 256, 0, stream>>>(deg, NN);

  for (int l = 0; l < 3; ++l) {
    cvt_bf16_kernel<<<16, 256, 0, stream>>>(Wl[l], wbf + l * 8192, DD * DD);
    cvt_bf16_kernel<<<16, 256, 0, stream>>>(Wr[l], wbf + l * 8192 + 4096, DD * DD);
  }

  const int tiles = NN / 16;                 // 6250
  const int gemm_blocks = (tiles + 7) / 8;   // 8 waves (tiles) per block

  const float* hin = x_raw;
  float* bufs[3] = {hA, hB, hA};
  for (int l = 0; l < 3; ++l) {
    zero_f32<<<2048, 256, 0, stream>>>(agg, (long long)NN * DD);
    scatter_kernel<<<4096, 256, 0, stream>>>(src, dst, hin, agg, (long long)EE * DD);
    sage_gemm_norm_kernel<<<gemm_blocks, 256, 0, stream>>>(
        agg, deg, hin, wbf + l * 8192, wbf + l * 8192 + 4096, bl[l], bufs[l], NN);
    hin = bufs[l];
  }

  readout_kernel<<<1024, 256, 0, stream>>>(hin, batch, (float*)d_out, (long long)NN * DD);
}
